// CohereAttention_6253472383326
// MI455X (gfx1250) — matrix operand order repaired and checked by reading, not compiled, not run on hardware
//
#include <hip/hip_runtime.h>

// ---------------------------------------------------------------------------
// CohereAttention for MI455X (gfx1250): bf16 WMMA pipeline + async LDS staging
//   B=2, S=2048, H=4096, NQ=32, NKV=8, D=128, THETA=10000
// ---------------------------------------------------------------------------

typedef __attribute__((ext_vector_type(16))) __bf16 bf16x16;
typedef __attribute__((ext_vector_type(8)))  float  f32x8;
typedef __attribute__((ext_vector_type(4)))  float  fx4;
typedef __attribute__((ext_vector_type(4)))  int    i32x4;

union FragAB { bf16x16 v; fx4 q[2]; };
union FragC  { f32x8  v; float f[8]; };
union Pack8  { fx4 v; __bf16 h[8]; };

__device__ __forceinline__ f32x8 wmma_bf16(bf16x16 a, bf16x16 b, f32x8 c) {
    return __builtin_amdgcn_wmma_f32_16x16x32_bf16(
        false, a, false, b, (short)0, c, false, false);
}

// ---- CDNA5 async global->LDS (ASYNCcnt) with safe fallback -----------------
#if __has_builtin(__builtin_amdgcn_global_load_async_to_lds_b128) && \
    __has_builtin(__builtin_amdgcn_s_wait_asynccnt)
#define HAS_ASYNC_LDS 1
typedef __attribute__((address_space(1))) i32x4 as1_i32x4;
typedef __attribute__((address_space(3))) i32x4 as3_i32x4;
#else
#define HAS_ASYNC_LDS 0
#endif

// copy 8 bf16 (16 bytes), both pointers 16B aligned
__device__ __forceinline__ void ld_g2l_b128(const __bf16* g, __bf16* l) {
#if HAS_ASYNC_LDS
    // generic->AS1: same address value; generic shared ->AS3: low 32 bits are
    // the LDS offset — int-casts avoid any
    // disjoint-address-space cast diagnostics.
    as1_i32x4* gp = (as1_i32x4*)(unsigned long long)g;
    as3_i32x4* lp = (as3_i32x4*)(unsigned int)(unsigned long long)l;
    __builtin_amdgcn_global_load_async_to_lds_b128(gp, lp, 0, 0);
#else
    *(fx4*)l = *(const fx4*)g;
#endif
}
__device__ __forceinline__ void g2l_wait() {
#if HAS_ASYNC_LDS
    __builtin_amdgcn_s_wait_asynccnt(0);
#endif
}

// ---------------------------------------------------------------------------
// f32 -> bf16 conversion (vectorized x4, grid-stride)
// ---------------------------------------------------------------------------
__global__ void cvt_f32_bf16(const float* __restrict__ s, __bf16* __restrict__ d,
                             long n) {
    long i = ((long)blockIdx.x * blockDim.x + threadIdx.x) * 4;
    long stride = (long)gridDim.x * blockDim.x * 4;
    for (; i < n; i += stride) {
        fx4 v = *(const fx4*)(s + i);
        d[i + 0] = (__bf16)v[0];
        d[i + 1] = (__bf16)v[1];
        d[i + 2] = (__bf16)v[2];
        d[i + 3] = (__bf16)v[3];
    }
}

// ---------------------------------------------------------------------------
// GEMM1: qkv = hidden_bf16[4096x4096] @ w_qkv_bf16[4096x6144]
// fused GPT-J RoPE epilogue -> Q[B,NQ,S,D], K[B,NKV,S,D], V[B,NKV,S,D] (bf16)
// block tile 128x128, K-step 32, 8 waves of 64(M)x32(N)
// ---------------------------------------------------------------------------
#define SA 40   // LDS stride (bf16 elems) for A tile [128][32]
#define SB 40   // LDS stride for transposed B tile [128 n][32 k]
#define SC 132  // LDS stride (f32) for C tile [128][128]

__global__ __launch_bounds__(256) void gemm_qkv_rope(
    const __bf16* __restrict__ A,     // hidden bf16 [4096][4096]
    const __bf16* __restrict__ Bw,    // w_qkv bf16 [4096][6144]
    const int*    __restrict__ positions,  // [B*S]
    __bf16* __restrict__ Qb, __bf16* __restrict__ Kb, __bf16* __restrict__ Vb)
{
    __shared__ __align__(16) __bf16 Al[128 * SA];
    __shared__ __align__(16) __bf16 Bl[128 * SB];
    __shared__ __align__(16) float  Cl[128 * SC];

    const int t    = threadIdx.x;
    const int lane = t & 31;
    const int wv   = t >> 5;
    const int l    = lane & 15;
    const int h    = lane >> 4;
    const int wm   = wv >> 2;        // 0..1
    const int wn   = wv & 3;         // 0..3
    const int mb   = blockIdx.y * 128;
    const int nb   = blockIdx.x * 128;
    const int K    = 4096, Nw = 6144;

    f32x8 acc[4][2];
#pragma unroll
    for (int i = 0; i < 4; ++i)
#pragma unroll
        for (int j = 0; j < 2; ++j) acc[i][j] = (f32x8)0.0f;

    for (int kb = 0; kb < K; kb += 32) {
        // A tile: [128 m][32 k] straight copy (async global->LDS)
        {
            int m = t >> 1, kh = (t & 1) * 16;
            const __bf16* g = A + (long)(mb + m) * K + kb + kh;
            ld_g2l_b128(g,     &Al[m * SA + kh]);
            ld_g2l_b128(g + 8, &Al[m * SA + kh + 8]);
        }
        // B tile: transpose to [128 n][32 k] (manual: scatter 2B elements)
#pragma unroll
        for (int it = 0; it < 2; ++it) {
            int task = t + it * 256;
            int k = task >> 4, n8 = (task & 15) * 8;
            Pack8 u;
            u.v = *(const fx4*)(Bw + (long)(kb + k) * Nw + nb + n8);
#pragma unroll
            for (int i = 0; i < 8; ++i) Bl[(n8 + i) * SB + k] = u.h[i];
        }
        g2l_wait();
        __syncthreads();

        bf16x16 af[4], bfr[2];
#pragma unroll
        for (int i = 0; i < 4; ++i) {
            const __bf16* p = &Al[(wm * 64 + i * 16 + l) * SA + h * 8];
            FragAB f; f.q[0] = *(const fx4*)p; f.q[1] = *(const fx4*)(p + 16);
            af[i] = f.v;
        }
#pragma unroll
        for (int j = 0; j < 2; ++j) {
            const __bf16* p = &Bl[(wn * 32 + j * 16 + l) * SB + h * 8];
            FragAB f; f.q[0] = *(const fx4*)p; f.q[1] = *(const fx4*)(p + 16);
            bfr[j] = f.v;
        }
#pragma unroll
        for (int i = 0; i < 4; ++i)
#pragma unroll
            for (int j = 0; j < 2; ++j)
                acc[i][j] = wmma_bf16(af[i], bfr[j], acc[i][j]);
        __syncthreads();
    }

    // accumulators -> Cl (f32)
#pragma unroll
    for (int i = 0; i < 4; ++i)
#pragma unroll
        for (int j = 0; j < 2; ++j) {
            FragC a; a.v = acc[i][j];
#pragma unroll
            for (int r = 0; r < 8; ++r) {
                int row = wm * 64 + i * 16 + r + 8 * h;
                int col = wn * 32 + j * 16 + l;
                Cl[row * SC + col] = a.f[r];
            }
        }
    __syncthreads();

    // RoPE epilogue: nb is 128-aligned -> whole tile is one head
    // 128 rows x 64 pairs = 8192 tasks, 32 per thread
    for (int it = 0; it < 32; ++it) {
        int id  = t + it * 256;
        int row = id >> 6;
        int j   = id & 63;
        int gm  = mb + row;
        int b   = gm >> 11;        // S = 2048
        int s   = gm & 2047;
        float x1 = Cl[row * SC + 2 * j];
        float x2 = Cl[row * SC + 2 * j + 1];
        if (nb < 4096) {           // Q region
            int head = nb >> 7;
            float invf = __expf(-((float)(2 * j) * (1.0f / 128.0f)) * 9.210340372f);
            float ang  = (float)positions[gm] * invf;
            float cs, sn; __sincosf(ang, &sn, &cs);
            __bf16* dst = Qb + ((long)(b * 32 + head) * 2048 + s) * 128;
            dst[2 * j]     = (__bf16)(x1 * cs - x2 * sn);
            dst[2 * j + 1] = (__bf16)(x2 * cs + x1 * sn);
        } else if (nb < 5120) {    // K region
            int head = (nb - 4096) >> 7;
            float invf = __expf(-((float)(2 * j) * (1.0f / 128.0f)) * 9.210340372f);
            float ang  = (float)positions[gm] * invf;
            float cs, sn; __sincosf(ang, &sn, &cs);
            __bf16* dst = Kb + ((long)(b * 8 + head) * 2048 + s) * 128;
            dst[2 * j]     = (__bf16)(x1 * cs - x2 * sn);
            dst[2 * j + 1] = (__bf16)(x2 * cs + x1 * sn);
        } else {                   // V region (no rope)
            int head = (nb - 5120) >> 7;
            __bf16* dst = Vb + ((long)(b * 8 + head) * 2048 + s) * 128;
            dst[2 * j]     = (__bf16)x1;
            dst[2 * j + 1] = (__bf16)x2;
        }
    }
}

// ---------------------------------------------------------------------------
// Flash attention: per block = (64 q rows, head hq, batch b); 4 waves x 16 rows
// KV tile = 32; causal; online softmax in f32; S and P*V via bf16 WMMA
// ---------------------------------------------------------------------------
#define SKT 136  // Kt [32 kv][128 d]
#define SVT 40   // Vt [128 d][32 kv]
#define SPT 40   // Ps [4*16 rows][32 kv]

__global__ __launch_bounds__(128) void flash_attn(
    const __bf16* __restrict__ Qb, const __bf16* __restrict__ Kb,
    const __bf16* __restrict__ Vb, __bf16* __restrict__ attn)
{
    __shared__ __align__(16) __bf16 Kt[32 * SKT];
    __shared__ __align__(16) __bf16 Vt[128 * SVT];
    __shared__ __align__(16) __bf16 Ps[64 * SPT];

    const int t    = threadIdx.x;
    const int lane = t & 31;
    const int wv   = t >> 5;
    const int l    = lane & 15;
    const int h    = lane >> 4;
    const int qb0  = blockIdx.x * 64;
    const int hq   = blockIdx.y;
    const int b    = blockIdx.z;
    const int hkv  = hq >> 2;
    const int S = 2048, D = 128;
    const float scale = 0.08838834764831845f;  // 128^-0.5

    const __bf16* Qh = Qb + (long)(b * 32 + hq)  * S * D;
    const __bf16* Kh = Kb + (long)(b * 8  + hkv) * S * D;
    const __bf16* Vh = Vb + (long)(b * 8  + hkv) * S * D;

    const int qr = qb0 + wv * 16;

    // Q fragments (A-layout), K-chunks of 32 over D=128
    bf16x16 qf[4];
#pragma unroll
    for (int ck = 0; ck < 4; ++ck) {
        const __bf16* p = Qh + (long)(qr + l) * D + ck * 32 + h * 8;
        FragAB f; f.q[0] = *(const fx4*)p; f.q[1] = *(const fx4*)(p + 16);
        qf[ck] = f.v;
    }

    f32x8 of[8];
#pragma unroll
    for (int i = 0; i < 8; ++i) of[i] = (f32x8)0.0f;
    float mrow[8], lrow[8];
#pragma unroll
    for (int r = 0; r < 8; ++r) { mrow[r] = -1e30f; lrow[r] = 0.0f; }

    const int kvend = qb0 + 64;
    for (int kvb = 0; kvb < kvend; kvb += 32) {
        __syncthreads();
        // K tile -> Kt[kv][d] via async b128 (B-operand ready for Q*K^T)
        {
            int kv = t >> 2, d32 = (t & 3) * 32;
            const __bf16* g = Kh + (long)(kvb + kv) * D + d32;
#pragma unroll
            for (int ii = 0; ii < 4; ++ii)
                ld_g2l_b128(g + ii * 8, &Kt[kv * SKT + d32 + ii * 8]);
        }
        // V tile -> Vt[d][kv] (transposed; B-operand ready for P*V)
        {
            int kv = t & 31, seg = t >> 5;
            const __bf16* g = Vh + (long)(kvb + kv) * D + seg * 32;
#pragma unroll
            for (int ii = 0; ii < 4; ++ii) {
                Pack8 u; u.v = *(const fx4*)(g + ii * 8);
                int d0 = seg * 32 + ii * 8;
#pragma unroll
                for (int i2 = 0; i2 < 8; ++i2) Vt[(d0 + i2) * SVT + kv] = u.h[i2];
            }
        }
        g2l_wait();
        __syncthreads();

        // S = Q * K^T   (16 q rows x 32 kv cols per wave)
        f32x8 sf[2];
        sf[0] = (f32x8)0.0f; sf[1] = (f32x8)0.0f;
#pragma unroll
        for (int j = 0; j < 2; ++j)
#pragma unroll
            for (int ck = 0; ck < 4; ++ck) {
                const __bf16* p = &Kt[(j * 16 + l) * SKT + ck * 32 + h * 8];
                FragAB f; f.q[0] = *(const fx4*)p; f.q[1] = *(const fx4*)(p + 16);
                sf[j] = wmma_bf16(qf[ck], f.v, sf[j]);
            }

        // scale + causal mask + row max (xor shuffles stay inside 16-lane half)
        FragC a0, a1; a0.v = sf[0]; a1.v = sf[1];
        float sv0[8], sv1[8], rmax[8];
#pragma unroll
        for (int r = 0; r < 8; ++r) {
            int grow = qr + r + 8 * h;
            float v0 = a0.f[r] * scale; if (kvb + l      > grow) v0 = -1e30f;
            float v1 = a1.f[r] * scale; if (kvb + 16 + l > grow) v1 = -1e30f;
            sv0[r] = v0; sv1[r] = v1;
            float m = fmaxf(v0, v1);
            m = fmaxf(m, __shfl_xor(m, 1));
            m = fmaxf(m, __shfl_xor(m, 2));
            m = fmaxf(m, __shfl_xor(m, 4));
            m = fmaxf(m, __shfl_xor(m, 8));
            rmax[r] = m;
        }
        // online softmax update + write P (bf16) to per-wave LDS patch
        float corr[8];
#pragma unroll
        for (int r = 0; r < 8; ++r) {
            float mnew = fmaxf(mrow[r], rmax[r]);
            corr[r] = __expf(mrow[r] - mnew);
            mrow[r] = mnew;
            float p0 = __expf(sv0[r] - mnew);
            float p1 = __expf(sv1[r] - mnew);
            float rs = p0 + p1;
            rs += __shfl_xor(rs, 1);
            rs += __shfl_xor(rs, 2);
            rs += __shfl_xor(rs, 4);
            rs += __shfl_xor(rs, 8);
            lrow[r] = lrow[r] * corr[r] + rs;
            Ps[(wv * 16 + r + 8 * h) * SPT + l]      = (__bf16)p0;
            Ps[(wv * 16 + r + 8 * h) * SPT + 16 + l] = (__bf16)p1;
        }
        __syncthreads();  // uniform trip count across all waves

        // P (A-layout from LDS) and O += P*V
        bf16x16 pa;
        {
            const __bf16* p = &Ps[(wv * 16 + l) * SPT + h * 8];
            FragAB f; f.q[0] = *(const fx4*)p; f.q[1] = *(const fx4*)(p + 16);
            pa = f.v;
        }
#pragma unroll
        for (int tn = 0; tn < 8; ++tn) {
            FragC oa; oa.v = of[tn];
#pragma unroll
            for (int r = 0; r < 8; ++r) oa.f[r] *= corr[r];
            const __bf16* p = &Vt[(tn * 16 + l) * SVT + h * 8];
            FragAB f; f.q[0] = *(const fx4*)p; f.q[1] = *(const fx4*)(p + 16);
            of[tn] = wmma_bf16(pa, f.v, oa.v);
        }
    }

    // finalize: O / l -> attn bf16 [B*S][4096]
#pragma unroll
    for (int r = 0; r < 8; ++r) lrow[r] = 1.0f / lrow[r];
#pragma unroll
    for (int tn = 0; tn < 8; ++tn) {
        FragC oa; oa.v = of[tn];
#pragma unroll
        for (int r = 0; r < 8; ++r) {
            int srow = qr + r + 8 * h;
            long idx = ((long)(b * S + srow)) * 4096 + hq * 128 + tn * 16 + l;
            attn[idx] = (__bf16)(oa.f[r] * lrow[r]);
        }
    }
}

// ---------------------------------------------------------------------------
// GEMM2: out_f32[4096x4096] = attn_bf16[4096x4096] @ w_o_bf16[4096x4096]
// ---------------------------------------------------------------------------
__global__ __launch_bounds__(256) void gemm_out(
    const __bf16* __restrict__ A, const __bf16* __restrict__ Bw,
    float* __restrict__ Out)
{
    __shared__ __align__(16) __bf16 Al[128 * SA];
    __shared__ __align__(16) __bf16 Bl[128 * SB];

    const int t    = threadIdx.x;
    const int lane = t & 31;
    const int wv   = t >> 5;
    const int l    = lane & 15;
    const int h    = lane >> 4;
    const int wm   = wv >> 2;
    const int wn   = wv & 3;
    const int mb   = blockIdx.y * 128;
    const int nb   = blockIdx.x * 128;
    const int K = 4096, Nw = 4096;

    f32x8 acc[4][2];
#pragma unroll
    for (int i = 0; i < 4; ++i)
#pragma unroll
        for (int j = 0; j < 2; ++j) acc[i][j] = (f32x8)0.0f;

    for (int kb = 0; kb < K; kb += 32) {
        {
            int m = t >> 1, kh = (t & 1) * 16;
            const __bf16* g = A + (long)(mb + m) * K + kb + kh;
            ld_g2l_b128(g,     &Al[m * SA + kh]);
            ld_g2l_b128(g + 8, &Al[m * SA + kh + 8]);
        }
#pragma unroll
        for (int it = 0; it < 2; ++it) {
            int task = t + it * 256;
            int k = task >> 4, n8 = (task & 15) * 8;
            Pack8 u;
            u.v = *(const fx4*)(Bw + (long)(kb + k) * Nw + nb + n8);
#pragma unroll
            for (int i = 0; i < 8; ++i) Bl[(n8 + i) * SB + k] = u.h[i];
        }
        g2l_wait();
        __syncthreads();

        bf16x16 af[4], bfr[2];
#pragma unroll
        for (int i = 0; i < 4; ++i) {
            const __bf16* p = &Al[(wm * 64 + i * 16 + l) * SA + h * 8];
            FragAB f; f.q[0] = *(const fx4*)p; f.q[1] = *(const fx4*)(p + 16);
            af[i] = f.v;
        }
#pragma unroll
        for (int j = 0; j < 2; ++j) {
            const __bf16* p = &Bl[(wn * 32 + j * 16 + l) * SB + h * 8];
            FragAB f; f.q[0] = *(const fx4*)p; f.q[1] = *(const fx4*)(p + 16);
            bfr[j] = f.v;
        }
#pragma unroll
        for (int i = 0; i < 4; ++i)
#pragma unroll
            for (int j = 0; j < 2; ++j)
                acc[i][j] = wmma_bf16(af[i], bfr[j], acc[i][j]);
        __syncthreads();
    }

#pragma unroll
    for (int i = 0; i < 4; ++i)
#pragma unroll
        for (int j = 0; j < 2; ++j) {
            FragC a; a.v = acc[i][j];
#pragma unroll
            for (int r = 0; r < 8; ++r) {
                long gm = mb + wm * 64 + i * 16 + r + 8 * h;
                long gn = nb + wn * 32 + j * 16 + l;
                Out[gm * 4096 + gn] = a.f[r];
            }
        }
}

// ---------------------------------------------------------------------------
// launch
// ---------------------------------------------------------------------------
extern "C" void kernel_launch(void* const* d_in, const int* in_sizes, int n_in,
                              void* d_out, int out_size, void* d_ws, size_t ws_size,
                              hipStream_t stream) {
    (void)in_sizes; (void)n_in; (void)out_size; (void)ws_size;
    const float* hidden    = (const float*)d_in[0];
    const float* w_qkv     = (const float*)d_in[1];
    const float* w_o       = (const float*)d_in[2];
    const int*   positions = (const int*)d_in[3];

    char* ws = (char*)d_ws;
    __bf16* hb    = (__bf16*)(ws + 0L);            // 4096*4096*2 = 32 MiB
    __bf16* wqkvb = (__bf16*)(ws + 33554432L);     // 4096*6144*2 = 48 MiB
    __bf16* wob   = (__bf16*)(ws + 83886080L);     // 4096*4096*2 = 32 MiB
    __bf16* Qbuf  = (__bf16*)(ws + 117440512L);    // 2*32*2048*128*2 = 32 MiB
    __bf16* Kbuf  = (__bf16*)(ws + 150994944L);    // 2*8*2048*128*2  =  8 MiB
    __bf16* Vbuf  = (__bf16*)(ws + 159383552L);    //                  8 MiB
    __bf16* attnb = (__bf16*)(ws + 167772160L);    // 4096*4096*2 = 32 MiB

    cvt_f32_bf16<<<2048, 256, 0, stream>>>(hidden, hb,    4096L * 4096L);
    cvt_f32_bf16<<<2048, 256, 0, stream>>>(w_qkv,  wqkvb, 4096L * 6144L);
    cvt_f32_bf16<<<2048, 256, 0, stream>>>(w_o,    wob,   4096L * 4096L);

    gemm_qkv_rope<<<dim3(48, 32), 256, 0, stream>>>(hb, wqkvb, positions,
                                                    Qbuf, Kbuf, Vbuf);
    flash_attn<<<dim3(32, 32, 2), 128, 0, stream>>>(Qbuf, Kbuf, Vbuf, attnb);
    gemm_out<<<dim3(32, 32), 256, 0, stream>>>(attnb, wob, (float*)d_out);
}
